// SACLRBatchMix_68307159875726
// MI455X (gfx1250) — matrix operand order repaired and checked by reading, not compiled, not run on hardware
//
#include <hip/hip_runtime.h>

typedef __attribute__((ext_vector_type(16))) _Float16 v16h;
typedef __attribute__((ext_vector_type(8)))  _Float16 v8h;
typedef __attribute__((ext_vector_type(8)))  float    v8f;

#define B_SZ   4096
#define TWOB   8192
#define D_SZ   256
// exp(4d-4) = exp2((d-1) * 4*log2(e))
#define C4LOG2E 5.770780163555854f

#if __has_builtin(__builtin_amdgcn_exp2f)
#define EXP2(x) __builtin_amdgcn_exp2f(x)
#else
#define EXP2(x) exp2f(x)
#endif

union V16U { v16h v; v8h p[2]; };

// Async global->LDS copy of 32 bytes per thread (two B128 ops; the 24-bit
// IOFFSET is added to BOTH the LDS and global addresses per ISA §4.4, so a
// single LDS-address VGPR covers both halves). Tracked by ASYNCcnt.
#define ASYNC_CP32(ldsaddr, gaddr)                                         \
    asm volatile("global_load_async_to_lds_b128 %0, %1, off\n\t"           \
                 "global_load_async_to_lds_b128 %0, %1, off offset:16"     \
                 :: "v"(ldsaddr), "v"(gaddr) : "memory")

#define WAIT_ASYNC0() asm volatile("s_wait_asynccnt 0x0" ::: "memory")

// Low 32 bits of a generic LDS pointer == LDS byte address (ISA §10.2:
// flat->LDS mapping truncates to addr[31:0]).
__device__ __forceinline__ unsigned lds_addr32(const void* p) {
    return (unsigned)(uintptr_t)p;
}

#define WMMA_F16(A, Bv, C) \
    __builtin_amdgcn_wmma_f32_16x16x32_f16(false, (A), false, (Bv), (short)0, (C), false, false)

// ---------------------------------------------------------------------------
// Kernel 1: L2-normalize rows of feats (8192 x 256 f32) -> f16.  1 wave / row.
// ---------------------------------------------------------------------------
__global__ __launch_bounds__(256) void k_normalize(const float* __restrict__ feats,
                                                   _Float16* __restrict__ Fh) {
    const int wave = threadIdx.x >> 5;
    const int lane = threadIdx.x & 31;
    const int row  = blockIdx.x * 8 + wave;          // grid = 1024
    const float4* src = (const float4*)(feats + (size_t)row * D_SZ + lane * 8);
    float4 x0 = src[0], x1 = src[1];
    float s = x0.x*x0.x + x0.y*x0.y + x0.z*x0.z + x0.w*x0.w
            + x1.x*x1.x + x1.y*x1.y + x1.z*x1.z + x1.w*x1.w;
    #pragma unroll
    for (int m = 16; m >= 1; m >>= 1) s += __shfl_xor(s, m, 32);
    const float rn = rsqrtf(s);
    v8h o;
    o[0]=(_Float16)(x0.x*rn); o[1]=(_Float16)(x0.y*rn);
    o[2]=(_Float16)(x0.z*rn); o[3]=(_Float16)(x0.w*rn);
    o[4]=(_Float16)(x1.x*rn); o[5]=(_Float16)(x1.y*rn);
    o[6]=(_Float16)(x1.z*rn); o[7]=(_Float16)(x1.w*rn);
    *(v8h*)(Fh + (size_t)row * D_SZ + lane * 8) = o;
}

// ---------------------------------------------------------------------------
// Kernel 2: zero the 8192 row-sum accumulators.
// ---------------------------------------------------------------------------
__global__ __launch_bounds__(256) void k_zero(float* __restrict__ S) {
    S[blockIdx.x * 256 + threadIdx.x] = 0.0f;       // grid = 32
}

// ---------------------------------------------------------------------------
// Kernel 3: pos[i] = dot(Fh[i], Fh[i+B]).  1 wave / pair.
// ---------------------------------------------------------------------------
__global__ __launch_bounds__(256) void k_pos(const _Float16* __restrict__ Fh,
                                             float* __restrict__ posArr) {
    const int wave = threadIdx.x >> 5;
    const int lane = threadIdx.x & 31;
    const int i = blockIdx.x * 8 + wave;             // grid = 512
    const v8h a = *(const v8h*)(Fh + (size_t)i * D_SZ + lane * 8);
    const v8h b = *(const v8h*)(Fh + (size_t)(i + B_SZ) * D_SZ + lane * 8);
    float s = 0.f;
    #pragma unroll
    for (int e = 0; e < 8; ++e) s += (float)a[e] * (float)b[e];
    #pragma unroll
    for (int m = 16; m >= 1; m >>= 1) s += __shfl_xor(s, m, 32);
    if (lane == 0) posArr[i] = s;
}

// ---------------------------------------------------------------------------
// Kernel 4: fused Gram + exp row-sum.
//   grid = (32 row-bands of 256) x (16 col-chunks of 32 tiles), 8 waves/block.
//   Each wave owns TWO 16-row M-tiles (A fragments for 32 rows register-
//   resident, all K=256) so every LDS B fragment feeds two WMMAs (16 WMMA :
//   16 ds_load_b128 per tile).  B tiles stream global->LDS with
//   GLOBAL_LOAD_ASYNC_TO_LDS_B128 into alternating LDS buffers (DMA of tile
//   j+1 overlaps the WMMAs of tile j), and B fragments double-buffer through
//   registers inside the K-loop so ds_loads overlap WMMA issue.
// ---------------------------------------------------------------------------
#define BPAD 264   // padded LDS row stride in halves (528B -> 4-dword bank skew)

__global__ __launch_bounds__(256) void k_gram_expsum(const _Float16* __restrict__ Fh,
                                                     float* __restrict__ S) {
    __shared__ _Float16 sB[2][16 * BPAD];

    const int tid  = threadIdx.x;
    const int wave = tid >> 5;
    const int lane = tid & 31;
    const int h    = lane >> 4;       // half-wave select
    const int l15  = lane & 15;
    const int rowBase = blockIdx.x * 256 + wave * 32;   // two M-tiles per wave

    // --- preload A fragments (contiguous k = 16h + s order, same as B), resident all K ---
    v16h a0[8], a1[8];
    {
        const _Float16* ar0 = Fh + (size_t)(rowBase + l15) * D_SZ;
        const _Float16* ar1 = Fh + (size_t)(rowBase + 16 + l15) * D_SZ;
        #pragma unroll
        for (int k = 0; k < 8; ++k) {
            const int base = k * 32 + 16 * h;
            V16U t0, t1;
            t0.p[0] = *(const v8h*)(ar0 + base);        // K = 32k+16h .. +7
            t0.p[1] = *(const v8h*)(ar0 + base + 8);    // K = 32k+16h+8 .. +7
            t1.p[0] = *(const v8h*)(ar1 + base);
            t1.p[1] = *(const v8h*)(ar1 + base + 8);
            a0[k] = t0.v;
            a1[k] = t1.v;
        }
    }

    float rs0[8] = {0.f,0.f,0.f,0.f,0.f,0.f,0.f,0.f};
    float rs1[8] = {0.f,0.f,0.f,0.f,0.f,0.f,0.f,0.f};

    // B-tile streaming: thread copies 32B of row (colBase+br), segment bs.
    const int br = tid >> 4;
    const int bs = tid & 15;
    const int jt0 = blockIdx.y * 32;  // 32 column tiles per chunk

    const _Float16* gp = Fh + (size_t)(jt0 * 16 + br) * D_SZ + bs * 16;
    const unsigned ldsDst0 = lds_addr32(&sB[0][br * BPAD + bs * 16]);
    const unsigned ldsDst1 = lds_addr32(&sB[1][br * BPAD + bs * 16]);
    const _Float16* fragB0 = &sB[0][l15 * BPAD + 16 * h];
    const _Float16* fragB1 = &sB[1][l15 * BPAD + 16 * h];

    // prologue: fill buffer 0
    ASYNC_CP32(ldsDst0, gp);
    gp += 16 * D_SZ;

    for (int jj = 0; jj < 32; ++jj) {
        const int cur = jj & 1;
        WAIT_ASYNC0();                 // my writes into buf[cur] are done
        __syncthreads();               // everyone's writes done; buf[1-cur] free
        if (jj + 1 < 32) {             // start DMA of next tile into buf[1-cur]
            ASYNC_CP32(cur ? ldsDst0 : ldsDst1, gp);
            gp += 16 * D_SZ;
        }

        const _Float16* bbase = cur ? fragB1 : fragB0;
        v8f acc0 = {}, acc1 = {};
        // register-double-buffered B fragments: load k+1 while WMMA k issues
        V16U tb0, tb1;
        tb0.p[0] = *(const v8h*)(bbase + 0);
        tb0.p[1] = *(const v8h*)(bbase + 8);
        #pragma unroll
        for (int k = 0; k < 8; ++k) {
            if (k & 1) {
                if (k < 7) {
                    tb0.p[0] = *(const v8h*)(bbase + (k + 1) * 32);
                    tb0.p[1] = *(const v8h*)(bbase + (k + 1) * 32 + 8);
                }
                acc0 = WMMA_F16(a0[k], tb1.v, acc0);
                acc1 = WMMA_F16(a1[k], tb1.v, acc1);
            } else {
                tb1.p[0] = *(const v8h*)(bbase + (k + 1) * 32);
                tb1.p[1] = *(const v8h*)(bbase + (k + 1) * 32 + 8);
                acc0 = WMMA_F16(a0[k], tb0.v, acc0);
                acc1 = WMMA_F16(a1[k], tb0.v, acc1);
            }
        }
        // fused exp(4d-4) row-sum accumulation
        #pragma unroll
        for (int e = 0; e < 8; ++e) {
            rs0[e] += EXP2((acc0[e] - 1.0f) * C4LOG2E);
            rs1[e] += EXP2((acc1[e] - 1.0f) * C4LOG2E);
        }
    }

    // reduce across the 16 lanes of each half (row m = e + 8h), then atomic
    #pragma unroll
    for (int e = 0; e < 8; ++e) {
        float v0 = rs0[e];
        v0 += __shfl_xor(v0, 1, 32);
        v0 += __shfl_xor(v0, 2, 32);
        v0 += __shfl_xor(v0, 4, 32);
        v0 += __shfl_xor(v0, 8, 32);
        float v1 = rs1[e];
        v1 += __shfl_xor(v1, 1, 32);
        v1 += __shfl_xor(v1, 2, 32);
        v1 += __shfl_xor(v1, 4, 32);
        v1 += __shfl_xor(v1, 8, 32);
        if (l15 == 0) {
            atomicAdd(&S[rowBase + e + 8 * h], v0);
            atomicAdd(&S[rowBase + 16 + e + 8 * h], v1);
        }
    }
}

// ---------------------------------------------------------------------------
// Kernel 5: scalar loss.
//   Zi = S[r] - 1 - exp(4*pos-4);  rep = Zi/(mu*Zhat+(1-mu)*Zi);
//   attr = 4 - 4*pos;  loss = sum(attr+rep)/(2B).
// ---------------------------------------------------------------------------
__global__ __launch_bounds__(256) void k_finalize(const float* __restrict__ S,
                                                  const float* __restrict__ posArr,
                                                  const int* __restrict__ idx,
                                                  const float* __restrict__ s_inv,
                                                  float* __restrict__ out) {
    __shared__ float red[256];
    float part = 0.f;
    for (int r = threadIdx.x; r < TWOB; r += 256) {
        const int i = r & (B_SZ - 1);
        const float pos = posArr[i];
        const float p   = EXP2((pos - 1.0f) * C4LOG2E);
        const float Zi  = S[r] - 1.0f - p;
        const float Zhat = s_inv[idx[i]] * ((2.0f * B_SZ - 2.0f) / (1.0e6f * 1.0e6f));
        const float Z   = 0.5f * Zhat + 0.5f * Zi;
        part += (4.0f - 4.0f * pos) + Zi / Z;
    }
    red[threadIdx.x] = part;
    __syncthreads();
    for (int s = 128; s > 0; s >>= 1) {
        if (threadIdx.x < s) red[threadIdx.x] += red[threadIdx.x + s];
        __syncthreads();
    }
    if (threadIdx.x == 0) out[0] = red[0] / (float)TWOB;
}

// ---------------------------------------------------------------------------
extern "C" void kernel_launch(void* const* d_in, const int* in_sizes, int n_in,
                              void* d_out, int out_size, void* d_ws, size_t ws_size,
                              hipStream_t stream) {
    const float* feats     = (const float*)d_in[0];
    const int*   feats_idx = (const int*)d_in[1];
    const float* s_inv     = (const float*)d_in[2];
    float* out = (float*)d_out;

    _Float16* Fh   = (_Float16*)d_ws;                                  // 4 MB
    float*    S    = (float*)((char*)d_ws + (size_t)TWOB * D_SZ * 2);  // 32 KB
    float*    posA = S + TWOB;                                         // 16 KB

    k_normalize  <<<dim3(1024),      256, 0, stream>>>(feats, Fh);
    k_zero       <<<dim3(32),        256, 0, stream>>>(S);
    k_pos        <<<dim3(512),       256, 0, stream>>>(Fh, posA);
    k_gram_expsum<<<dim3(32, 16),    256, 0, stream>>>(Fh, S);
    k_finalize   <<<dim3(1),         256, 0, stream>>>(S, posA, feats_idx, s_inv, out);
}
